// SparseMoeBlockWithLinearExperts_83176336654412
// MI455X (gfx1250) — matrix operand order repaired and checked by reading, not compiled, not run on hardware
//
#include <hip/hip_runtime.h>
#include <hip/hip_bf16.h>
#include <hip/hip_fp16.h>

typedef __attribute__((ext_vector_type(16))) _Float16 v16h;
typedef __attribute__((ext_vector_type(8)))  _Float16 v8h;
typedef __attribute__((ext_vector_type(4)))  _Float16 v4h;
typedef __attribute__((ext_vector_type(8)))  float    v8f;
typedef __attribute__((ext_vector_type(4)))  float    v4f;

#define HDIM 1024
#define IDIM 2048
#define NEXP 8
#define BM 128
#define BN 64
#define BK 32

__device__ __forceinline__ int imin(int a, int b) { return a < b ? a : b; }

// ---- CDNA5 async global->LDS copy (ASYNCcnt-tracked, no VGPR round trip) ----
__device__ __forceinline__ void async_copy_b128(unsigned int lds_off, const void* gptr)
{
    asm volatile("global_load_async_to_lds_b128 %0, %1, off"
                 :: "v"(lds_off), "v"((unsigned long long)(uintptr_t)gptr)
                 : "memory");
}

template <int N>
__device__ __forceinline__ void wait_asynccnt()
{
    asm volatile("s_wait_asynccnt %0" :: "i"(N) : "memory");
}

__device__ __forceinline__ unsigned int lds_addr(const void* p)
{
    // addrspace(3) flat address: low 32 bits are the LDS byte offset
    return (unsigned int)(uintptr_t)p;
}

// silu via hardware v_rcp_f32 (denominator >= 1, approx is safe here)
__device__ __forceinline__ float silu_fast(float g)
{
    return g * __builtin_amdgcn_rcpf(1.0f + __expf(-g));
}

// ---------------- router: wave-per-token, top-2 of 8 experts ----------------
__global__ __launch_bounds__(256) void router_kernel(
    const float* __restrict__ x, const float* __restrict__ rw,
    int* __restrict__ counts, int* __restrict__ tokList,
    float* __restrict__ wList, int N)
{
    int gid  = blockIdx.x * blockDim.x + threadIdx.x;
    int tok  = gid >> 5;          // one wave32 per token
    int lane = gid & 31;
    if (tok >= N) return;

    const float* xr = x + (size_t)tok * HDIM;
    float acc[NEXP];
#pragma unroll
    for (int e = 0; e < NEXP; ++e) acc[e] = 0.f;

    for (int h = lane; h < HDIM; h += 32) {
        float xv = xr[h];
#pragma unroll
        for (int e = 0; e < NEXP; ++e)
            acc[e] = fmaf(xv, rw[e * HDIM + h], acc[e]);
    }
#pragma unroll
    for (int e = 0; e < NEXP; ++e)
        for (int off = 16; off > 0; off >>= 1)
            acc[e] += __shfl_xor(acc[e], off, 32);

    if (lane == 0) {
        float mx = acc[0];
#pragma unroll
        for (int e = 1; e < NEXP; ++e) mx = fmaxf(mx, acc[e]);
        float p[NEXP];
#pragma unroll
        for (int e = 0; e < NEXP; ++e) p[e] = __expf(acc[e] - mx);
        int i0 = 0;
#pragma unroll
        for (int e = 1; e < NEXP; ++e) if (p[e] > p[i0]) i0 = e;
        int i1 = -1;
#pragma unroll
        for (int e = 0; e < NEXP; ++e)
            if (e != i0 && (i1 < 0 || p[e] > p[i1])) i1 = e;
        float w0 = p[i0], w1 = p[i1];
        float inv = 1.0f / (w0 + w1);   // renormalized top-2 (softmax denom cancels)
        w0 *= inv; w1 *= inv;
        int pos0 = atomicAdd(&counts[i0], 1);
        tokList[(size_t)i0 * N + pos0] = tok;
        wList [(size_t)i0 * N + pos0] = w0;
        int pos1 = atomicAdd(&counts[i1], 1);
        tokList[(size_t)i1 * N + pos1] = tok;
        wList [(size_t)i1 * N + pos1] = w1;
    }
}

// ---------------- tiny exclusive scan over 8 expert counts ----------------
__global__ void scan_kernel(const int* __restrict__ counts, int* __restrict__ bases)
{
    if (blockIdx.x == 0 && threadIdx.x == 0) {
        int b = 0;
        for (int e = 0; e < NEXP; ++e) { bases[e] = b; b += counts[e]; }
        bases[NEXP] = b;
    }
}

// ---------------- f32 -> f16 activation conversion ----------------
__global__ __launch_bounds__(256) void cvt_kernel(
    const float* __restrict__ x, _Float16* __restrict__ y, size_t n)
{
    size_t i = ((size_t)blockIdx.x * blockDim.x + threadIdx.x) * 4;
    if (i + 3 < n) {
        v4f v = *(const v4f*)(x + i);
        v4h o;
#pragma unroll
        for (int j = 0; j < 4; ++j) o[j] = (_Float16)v[j];
        *(v4h*)(y + i) = o;
    }
}

// ---------------- GEMM1: fused gate+up SwiGLU, gathered rows ----------------
// Double-buffered software pipeline:
//   stage k+1: async X tile (global->LDS), weight f32 tiles (global->VGPR)
//   stage k  : WMMA on LDS buffers
//   tail     : convert staged weights -> LDS buf^1
__global__ __launch_bounds__(256) void moe_gemm1(
    const _Float16* __restrict__ X16,
    const float* __restrict__ gate_w,
    const float* __restrict__ up_w,
    const int* __restrict__ counts,
    const int* __restrict__ bases,
    const int* __restrict__ tokList,
    _Float16* __restrict__ Abuf,
    int N)
{
    const int e  = blockIdx.z;
    const int Ne = counts[e];
    const int m0 = blockIdx.x * BM;
    if (m0 >= Ne) return;
    const int n0 = blockIdx.y * BN;
    const int slotBase = bases[e];
    const int* toks = tokList + (size_t)e * N;
    const float* Wg = gate_w + (size_t)e * IDIM * HDIM;
    const float* Wu = up_w   + (size_t)e * IDIM * HDIM;

    __shared__ _Float16 sX[2][BM][BK];   // 2 x 8 KB
    __shared__ _Float16 sG[2][BN][BK];   // 2 x 4 KB
    __shared__ _Float16 sU[2][BN][BK];   // 2 x 4 KB
    __shared__ int sTok[BM];

    const int tid  = threadIdx.x;
    const int lane = tid & 31;
    const int wid  = tid >> 5;        // 8 waves
    const int wm   = wid >> 2;        // 0..1 -> 64-row band
    const int wn   = wid & 3;         // 0..3 -> 16-col band

    if (tid < BM) {
        int m = m0 + tid;
        // clamp instead of zero-fill: async copy has no masking; out-of-range
        // rows produce garbage rows that are masked at the epilogue store.
        sTok[tid] = toks[imin(m, Ne - 1)];
    }
    __syncthreads();

    // per-thread staging geometry
    const int xrow = tid >> 1;
    const int xseg = (tid & 1) * 16;
    const int myTok = sTok[xrow];
    const _Float16* xbase = X16 + (size_t)myTok * HDIM + xseg;
    const unsigned int ldsX0 = lds_addr(&sX[0][xrow][xseg]);
    const unsigned int xBufStride = (unsigned int)(BM * BK * sizeof(_Float16));

    const int widx = tid * 8;
    const int wrow = widx >> 5;
    const int wk   = widx & 31;
    const float* gbase = Wg + (size_t)(n0 + wrow) * HDIM + wk;
    const float* ubase = Wu + (size_t)(n0 + wrow) * HDIM + wk;

    v8f cg[4] = {};
    v8f cu[4] = {};

    const int ar = lane & 15;
    const int ak = (lane >> 4) * 8;   // A frag: lanes 0-15 K[0..8)+[16..24), 16-31 K[8..16)+[24..32)
    const int bc = lane & 15;
    const int bk = (lane >> 4) * 8;   // B frag: lanes 0-15 K[0..8)+[16..24), 16-31 K[8..16)+[24..32)

    // ---- prologue: stage 0 ----
    async_copy_b128(ldsX0, xbase);
    async_copy_b128(ldsX0 + 16, xbase + 8);
    v4f g0 = *(const v4f*)(gbase);
    v4f g1 = *(const v4f*)(gbase + 4);
    v4f u0 = *(const v4f*)(ubase);
    v4f u1 = *(const v4f*)(ubase + 4);
#pragma unroll
    for (int j = 0; j < 4; ++j) {
        sG[0][wrow][wk + j]     = (_Float16)g0[j];
        sG[0][wrow][wk + 4 + j] = (_Float16)g1[j];
        sU[0][wrow][wk + j]     = (_Float16)u0[j];
        sU[0][wrow][wk + 4 + j] = (_Float16)u1[j];
    }

    int buf = 0;
    for (int k0 = 0; k0 < HDIM; k0 += BK) {
        const bool hasNext = (k0 + BK) < HDIM;
        if (hasNext) {
            const int nk = k0 + BK;
            unsigned int l = ldsX0 + (unsigned int)(buf ^ 1) * xBufStride;
            async_copy_b128(l, xbase + nk);
            async_copy_b128(l + 16, xbase + nk + 8);
            g0 = *(const v4f*)(gbase + nk);
            g1 = *(const v4f*)(gbase + nk + 4);
            u0 = *(const v4f*)(ubase + nk);
            u1 = *(const v4f*)(ubase + nk + 4);
            __builtin_prefetch(gbase + nk + BK, 0, 1);
            __builtin_prefetch(ubase + nk + BK, 0, 1);
            wait_asynccnt<2>();   // stage k done; stage k+1 still in flight
        } else {
            wait_asynccnt<0>();
        }
        __syncthreads();

        // ---- compute stage k on buf ----
        v16h bg, bu;
        {
            v8h lo = *(const v8h*)&sG[buf][wn * 16 + bc][bk];
            v8h hi = *(const v8h*)&sG[buf][wn * 16 + bc][bk + 16];
#pragma unroll
            for (int j = 0; j < 8; ++j) { bg[j] = lo[j]; bg[8 + j] = hi[j]; }
            lo = *(const v8h*)&sU[buf][wn * 16 + bc][bk];
            hi = *(const v8h*)&sU[buf][wn * 16 + bc][bk + 16];
#pragma unroll
            for (int j = 0; j < 8; ++j) { bu[j] = lo[j]; bu[8 + j] = hi[j]; }
        }
#pragma unroll
        for (int i = 0; i < 4; ++i) {
            int row = wm * 64 + i * 16 + ar;
            v8h lo = *(const v8h*)&sX[buf][row][ak];
            v8h hi = *(const v8h*)&sX[buf][row][ak + 16];
            v16h a;
#pragma unroll
            for (int j = 0; j < 8; ++j) { a[j] = lo[j]; a[8 + j] = hi[j]; }
            cg[i] = __builtin_amdgcn_wmma_f32_16x16x32_f16(
                false, a, false, bg, (short)0, cg[i], false, false);
            cu[i] = __builtin_amdgcn_wmma_f32_16x16x32_f16(
                false, a, false, bu, (short)0, cu[i], false, false);
        }
        __syncthreads();

        if (hasNext) {
#pragma unroll
            for (int j = 0; j < 4; ++j) {
                sG[buf ^ 1][wrow][wk + j]     = (_Float16)g0[j];
                sG[buf ^ 1][wrow][wk + 4 + j] = (_Float16)g1[j];
                sU[buf ^ 1][wrow][wk + j]     = (_Float16)u0[j];
                sU[buf ^ 1][wrow][wk + 4 + j] = (_Float16)u1[j];
            }
            buf ^= 1;
        }
    }

    // epilogue: silu(g) * u -> f16 slot-major intermediate
    const int mloc = (lane >> 4) * 8;
    const int col  = n0 + wn * 16 + (lane & 15);
#pragma unroll
    for (int i = 0; i < 4; ++i) {
#pragma unroll
        for (int r = 0; r < 8; ++r) {
            int m = m0 + wm * 64 + i * 16 + mloc + r;
            if (m < Ne) {
                float hv = silu_fast(cg[i][r]) * cu[i][r];
                Abuf[(size_t)(slotBase + m) * IDIM + col] = (_Float16)hv;
            }
        }
    }
}

// ---------------- GEMM2: down projection + weighted scatter-add ----------------
__global__ __launch_bounds__(256) void moe_gemm2(
    const _Float16* __restrict__ Abuf,
    const float* __restrict__ down_w,
    const int* __restrict__ counts,
    const int* __restrict__ bases,
    const int* __restrict__ tokList,
    const float* __restrict__ wList,
    float* __restrict__ out,
    int N)
{
    const int e  = blockIdx.z;
    const int Ne = counts[e];
    const int m0 = blockIdx.x * BM;
    if (m0 >= Ne) return;
    const int n0 = blockIdx.y * BN;     // column in H
    const int slotBase = bases[e];
    const float* Wd = down_w + (size_t)e * HDIM * IDIM;

    __shared__ _Float16 sA[2][BM][BK];
    __shared__ _Float16 sW[2][BN][BK];
    __shared__ int   sTok[BM];
    __shared__ float sRw[BM];

    const int tid  = threadIdx.x;
    const int lane = tid & 31;
    const int wid  = tid >> 5;
    const int wm   = wid >> 2;
    const int wn   = wid & 3;

    if (tid < BM) {
        int m = m0 + tid;
        sTok[tid] = (m < Ne) ? tokList[(size_t)e * N + m] : -1;
        sRw[tid]  = (m < Ne) ? wList[(size_t)e * N + m] : 0.f;
    }
    __syncthreads();

    const int xrow = tid >> 1;
    const int xseg = (tid & 1) * 16;
    const int arowClamped = imin(m0 + xrow, Ne - 1);
    const _Float16* abase = Abuf + (size_t)(slotBase + arowClamped) * IDIM + xseg;
    const unsigned int ldsA0 = lds_addr(&sA[0][xrow][xseg]);
    const unsigned int aBufStride = (unsigned int)(BM * BK * sizeof(_Float16));

    const int widx = tid * 8;
    const int wrow = widx >> 5;
    const int wk   = widx & 31;
    const float* dbase = Wd + (size_t)(n0 + wrow) * IDIM + wk;

    v8f cd[4] = {};

    const int ar = lane & 15;
    const int ak = (lane >> 4) * 8;
    const int bc = lane & 15;
    const int bk = (lane >> 4) * 8;

    // ---- prologue: stage 0 ----
    async_copy_b128(ldsA0, abase);
    async_copy_b128(ldsA0 + 16, abase + 8);
    v4f d0 = *(const v4f*)(dbase);
    v4f d1 = *(const v4f*)(dbase + 4);
#pragma unroll
    for (int j = 0; j < 4; ++j) {
        sW[0][wrow][wk + j]     = (_Float16)d0[j];
        sW[0][wrow][wk + 4 + j] = (_Float16)d1[j];
    }

    int buf = 0;
    for (int k0 = 0; k0 < IDIM; k0 += BK) {
        const bool hasNext = (k0 + BK) < IDIM;
        if (hasNext) {
            const int nk = k0 + BK;
            unsigned int l = ldsA0 + (unsigned int)(buf ^ 1) * aBufStride;
            async_copy_b128(l, abase + nk);
            async_copy_b128(l + 16, abase + nk + 8);
            d0 = *(const v4f*)(dbase + nk);
            d1 = *(const v4f*)(dbase + nk + 4);
            __builtin_prefetch(dbase + nk + BK, 0, 1);
            wait_asynccnt<2>();
        } else {
            wait_asynccnt<0>();
        }
        __syncthreads();

        v16h b;
        {
            v8h lo = *(const v8h*)&sW[buf][wn * 16 + bc][bk];
            v8h hi = *(const v8h*)&sW[buf][wn * 16 + bc][bk + 16];
#pragma unroll
            for (int j = 0; j < 8; ++j) { b[j] = lo[j]; b[8 + j] = hi[j]; }
        }
#pragma unroll
        for (int i = 0; i < 4; ++i) {
            int row = wm * 64 + i * 16 + ar;
            v8h lo = *(const v8h*)&sA[buf][row][ak];
            v8h hi = *(const v8h*)&sA[buf][row][ak + 16];
            v16h a;
#pragma unroll
            for (int j = 0; j < 8; ++j) { a[j] = lo[j]; a[8 + j] = hi[j]; }
            cd[i] = __builtin_amdgcn_wmma_f32_16x16x32_f16(
                false, a, false, b, (short)0, cd[i], false, false);
        }
        __syncthreads();

        if (hasNext) {
#pragma unroll
            for (int j = 0; j < 4; ++j) {
                sW[buf ^ 1][wrow][wk + j]     = (_Float16)d0[j];
                sW[buf ^ 1][wrow][wk + 4 + j] = (_Float16)d1[j];
            }
            buf ^= 1;
        }
    }

    const int mloc = (lane >> 4) * 8;
    const int col  = n0 + wn * 16 + (lane & 15);
#pragma unroll
    for (int i = 0; i < 4; ++i) {
#pragma unroll
        for (int r = 0; r < 8; ++r) {
            int m = m0 + wm * 64 + i * 16 + mloc + r;
            if (m < Ne) {
                int   tok = sTok[m - m0];
                float w   = sRw[m - m0];
                // hardware GLOBAL_ATOMIC_ADD_F32 (top-2 combine)
                unsafeAtomicAdd(&out[(size_t)tok * HDIM + col], w * cd[i][r]);
            }
        }
    }
}

// ---------------- host launch ----------------
extern "C" void kernel_launch(void* const* d_in, const int* in_sizes, int n_in,
                              void* d_out, int out_size, void* d_ws, size_t ws_size,
                              hipStream_t stream)
{
    const float* hidden  = (const float*)d_in[0];
    const float* routerW = (const float*)d_in[1];
    const float* gateW   = (const float*)d_in[2];
    const float* upW     = (const float*)d_in[3];
    const float* downW   = (const float*)d_in[4];
    float* out = (float*)d_out;

    const int N = in_sizes[0] / HDIM;   // 16384 tokens

    char* ws = (char*)d_ws;
    size_t off = 0;
    auto alloc = [&](size_t bytes) -> char* {
        char* p = ws + off;
        off = (off + bytes + 255) & ~(size_t)255;
        return p;
    };
    _Float16* X16  = (_Float16*)alloc((size_t)N * HDIM * sizeof(_Float16));
    int*      cnts = (int*)     alloc(NEXP * sizeof(int));
    int*      bas  = (int*)     alloc((NEXP + 1) * sizeof(int));
    int*      tlst = (int*)     alloc((size_t)NEXP * N * sizeof(int));
    float*    wlst = (float*)   alloc((size_t)NEXP * N * sizeof(float));
    _Float16* Abuf = (_Float16*)alloc((size_t)2 * N * IDIM * sizeof(_Float16));

    hipMemsetAsync(out,  0, (size_t)out_size * sizeof(float), stream);
    hipMemsetAsync(cnts, 0, NEXP * sizeof(int), stream);

    router_kernel<<<dim3((N * 32 + 255) / 256), 256, 0, stream>>>(
        hidden, routerW, cnts, tlst, wlst, N);
    scan_kernel<<<1, 32, 0, stream>>>(cnts, bas);
    {
        size_t nElems = (size_t)N * HDIM;
        int blocks = (int)((nElems / 4 + 255) / 256);
        cvt_kernel<<<blocks, 256, 0, stream>>>(hidden, X16, nElems);
    }
    moe_gemm1<<<dim3(N / BM, IDIM / BN, NEXP), 256, 0, stream>>>(
        X16, gateW, upW, cnts, bas, tlst, Abuf, N);
    moe_gemm2<<<dim3(N / BM, HDIM / BN, NEXP), 256, 0, stream>>>(
        Abuf, downW, cnts, bas, tlst, wlst, out, N);
}
